// EncoderBlock_1030792151496
// MI455X (gfx1250) — matrix operand order repaired and checked by reading, not compiled, not run on hardware
//
#include <hip/hip_runtime.h>

// ---------------------------------------------------------------------------
// EncoderBlock fused for MI455X (gfx1250, wave32, WMMA bf16 16x16x32)
// N=32 L=1024 C=3 KF=256 H=768 M=3072, T = N*L = 32768 tokens
// ---------------------------------------------------------------------------

#define NTHREADS 256          // 8 wave32 per workgroup
#define TOK_TILE 32           // tokens per workgroup (2 x 16-row WMMA tiles)
#define HDIM     768
#define KFDIM    256
#define QKVDIM   2304         // 3 * 768
#define MDIM     3072
#define TTOK     (32 * 1024)

typedef unsigned short ushort_t;
typedef __attribute__((ext_vector_type(16))) __bf16 v16bf;
typedef __attribute__((ext_vector_type(8)))  float  v8f;

__device__ __forceinline__ ushort_t f2bf(float f) {
  unsigned int u = __float_as_uint(f);
  u += 0x7FFFu + ((u >> 16) & 1u);          // RNE to bf16
  return (ushort_t)(u >> 16);
}
__device__ __forceinline__ float bf2f(ushort_t h) {
  return __uint_as_float(((unsigned int)h) << 16);
}

union Frag { v16bf v; uint4 q[2]; };

// A fragment (16x32 bf16, MxK) from an LDS row pointer already offset to
// (row m = lane&15, k-base). Lane-half 'hi' owns the contiguous K run
// [hi*16, hi*16+16) (matches B) -> two contiguous 16B reads.
__device__ __forceinline__ v16bf load_a_lds(const ushort_t* rowk, int hi) {
  Frag f;
  const ushort_t* p = rowk + hi * 16;
  f.q[0] = *(const uint4*)(p);
  f.q[1] = *(const uint4*)(p + 8);
  return f.v;
}
// B fragment (32x16 bf16, KxN) from [col][K]-transposed weights: the lane's
// 16 K-values are contiguous -> one 32B read.
__device__ __forceinline__ v16bf load_b16(const ushort_t* p) {
  Frag f;
  const uint4* q = (const uint4*)p;
  f.q[0] = q[0];
  f.q[1] = q[1];
  return f.v;
}

#define WMMA_BF16(a, b, c) \
  __builtin_amdgcn_wmma_f32_16x16x32_bf16(false, (a), false, (b), (short)0, (c), false, false)

// ---------------------------------------------------------------------------
// Weight prep: f32 [K][Ncols] row-major -> bf16 [Ncols][K] (B-fragment friendly)
// blockIdx.y = batch (per-channel W_qkv)
// ---------------------------------------------------------------------------
__global__ void k_convert_transpose(const float* __restrict__ in,
                                    ushort_t* __restrict__ out,
                                    int K, int Ncols) {
  size_t total = (size_t)K * (size_t)Ncols;
  size_t base  = (size_t)blockIdx.y * total;
  for (size_t e = (size_t)blockIdx.x * blockDim.x + threadIdx.x; e < total;
       e += (size_t)gridDim.x * blockDim.x) {
    size_t k = e / (size_t)Ncols;
    size_t n = e - k * (size_t)Ncols;
    out[base + n * (size_t)K + k] = f2bf(in[base + e]);
  }
}

// ---------------------------------------------------------------------------
// Kernel 1: LN1 -> per-channel QKV (WMMA) -> criss-cross attention -> out-proj
//           (WMMA) -> + x residual -> xr (f32, workspace)
// ---------------------------------------------------------------------------
__global__ __launch_bounds__(NTHREADS)
void k_ln1_qkv_attn(const float* __restrict__ x,
                    const float* __restrict__ b_qkv,
                    const float* __restrict__ b_out,
                    const float* __restrict__ ln1g,
                    const float* __restrict__ ln1b,
                    const ushort_t* __restrict__ wqkv_t,   // [C][768 cols][256 K]
                    const ushort_t* __restrict__ wout_t,   // [256 cols][256 K]
                    float* __restrict__ xr) {
  extern __shared__ unsigned char smem_raw[];
  float*    red1  = (float*)smem_raw;                 // 256
  float*    red2  = red1 + NTHREADS;                  // 256
  float*    muv   = red2 + NTHREADS;                  // 32
  float*    rsv   = muv + TOK_TILE;                   // 32
  ushort_t* hbf   = (ushort_t*)(rsv + TOK_TILE);      // [32][768]  bf16 LN1(x)
  ushort_t* qkvbf = hbf + TOK_TILE * HDIM;            // [32][2304] bf16 qkv
  ushort_t* ctxbf = qkvbf + TOK_TILE * QKVDIM;        // [32][768]  bf16 ctx

  const int tid = threadIdx.x;
  const size_t tile0 = (size_t)blockIdx.x * TOK_TILE;
  const float* xg = x + tile0 * HDIM;

  // ---- LN1 stats: 8 threads per token
  {
    int tok = tid >> 3, j = tid & 7;
    const float* xt = xg + (size_t)tok * HDIM;
    float s = 0.f, s2 = 0.f;
    for (int e = j; e < HDIM; e += 8) { float v = xt[e]; s += v; s2 += v * v; }
    red1[tid] = s; red2[tid] = s2;
  }
  __syncthreads();
  if (tid < TOK_TILE) {
    float s = 0.f, s2 = 0.f;
    for (int j = 0; j < 8; j++) { s += red1[tid * 8 + j]; s2 += red2[tid * 8 + j]; }
    float mu  = s * (1.f / HDIM);
    float var = s2 * (1.f / HDIM) - mu * mu;
    muv[tid] = mu;
    rsv[tid] = rsqrtf(var + 1e-6f);
  }
  __syncthreads();
  for (int e = tid; e < TOK_TILE * HDIM; e += NTHREADS) {
    int tok = e / HDIM, h = e - tok * HDIM;
    hbf[e] = f2bf((xg[e] - muv[tok]) * rsv[tok] * ln1g[h] + ln1b[h]);
  }
  __syncthreads();

  const int wave = tid >> 5, lane = tid & 31;
  const int hi = lane >> 4, lm = lane & 15;

  // ---- QKV: per channel, [32 tok x 256] @ [256 x 768], 144 col-tile jobs
  for (int job = wave; job < 3 * 48; job += 8) {
    int c = job / 48, ct = job - c * 48;
    int col = ct * 16 + lm;
    const ushort_t* bw = wqkv_t + ((size_t)c * HDIM + col) * KFDIM + hi * 16;
    const ushort_t* a0 = hbf + (size_t)lm * HDIM + c * KFDIM;
    const ushort_t* a1 = hbf + (size_t)(lm + 16) * HDIM + c * KFDIM;
    v8f acc0 = {}, acc1 = {};
    for (int kk = 0; kk < KFDIM; kk += 32) {
      __builtin_prefetch(bw + kk + 64, 0, 0);   // global_prefetch_b8
      v16bf bfrag = load_b16(bw + kk);
      acc0 = WMMA_BF16(load_a_lds(a0 + kk, hi), bfrag, acc0);
      acc1 = WMMA_BF16(load_a_lds(a1 + kk, hi), bfrag, acc1);
    }
    float bias = b_qkv[c * HDIM + col];
    ushort_t* q0 = qkvbf + c * HDIM + col;
    #pragma unroll
    for (int i = 0; i < 8; i++) {
      int r = i + 8 * hi;
      q0[(size_t)r * QKVDIM]        = f2bf(acc0[i] + bias);
      q0[(size_t)(r + 16) * QKVDIM] = f2bf(acc1[i] + bias);
    }
  }
  __syncthreads();

  // ---- criss-cross attention: per (token, kf), C=3 softmax over channels
  for (int e = tid; e < TOK_TILE * KFDIM; e += NTHREADS) {
    int tok = e >> 8, kf = e & 255;
    const ushort_t* qb = qkvbf + (size_t)tok * QKVDIM;
    float q[3], k[3], v[3];
    #pragma unroll
    for (int c = 0; c < 3; c++) {
      q[c] = bf2f(qb[c * HDIM + kf]);
      k[c] = bf2f(qb[c * HDIM + KFDIM + kf]);
      v[c] = bf2f(qb[c * HDIM + 2 * KFDIM + kf]);
    }
    float c0 = 0.f, c1 = 0.f, c2 = 0.f;
    const float scale = 0.0625f;                // 1/sqrt(KF)
    #pragma unroll
    for (int c = 0; c < 3; c++) {
      float s0 = q[c] * k[0] * scale, s1 = q[c] * k[1] * scale, s2 = q[c] * k[2] * scale;
      float m  = fmaxf(s0, fmaxf(s1, s2));
      float e0 = __expf(s0 - m), e1 = __expf(s1 - m), e2 = __expf(s2 - m);
      float w  = v[c] / (e0 + e1 + e2);
      c0 += e0 * w; c1 += e1 * w; c2 += e2 * w;
    }
    ushort_t* cb = ctxbf + (size_t)tok * HDIM + kf;
    cb[0] = f2bf(c0); cb[KFDIM] = f2bf(c1); cb[2 * KFDIM] = f2bf(c2);
  }
  __syncthreads();

  // ---- out projection: per channel [32 x 256] @ [256 x 256], + x residual
  for (int job = wave; job < 3 * 16; job += 8) {
    int c = job / 16, ct = job - c * 16;
    int col = ct * 16 + lm;
    const ushort_t* bw = wout_t + (size_t)col * KFDIM + hi * 16;
    const ushort_t* a0 = ctxbf + (size_t)lm * HDIM + c * KFDIM;
    const ushort_t* a1 = ctxbf + (size_t)(lm + 16) * HDIM + c * KFDIM;
    v8f acc0 = {}, acc1 = {};
    for (int kk = 0; kk < KFDIM; kk += 32) {
      v16bf bfrag = load_b16(bw + kk);
      acc0 = WMMA_BF16(load_a_lds(a0 + kk, hi), bfrag, acc0);
      acc1 = WMMA_BF16(load_a_lds(a1 + kk, hi), bfrag, acc1);
    }
    float bias = b_out[col];
    int hcol = c * KFDIM + col;
    #pragma unroll
    for (int i = 0; i < 8; i++) {
      int r = i + 8 * hi;
      size_t g0 = (tile0 + r) * HDIM + hcol;
      size_t g1 = (tile0 + r + 16) * HDIM + hcol;
      xr[g0] = acc0[i] + bias + x[g0];
      xr[g1] = acc1[i] + bias + x[g1];
    }
  }
}

// ---------------------------------------------------------------------------
// Kernel 2: LN2 -> GEMM1 (768->3072) + exact GELU -> GEMM2 (3072->768) -> +xr
// ---------------------------------------------------------------------------
__global__ __launch_bounds__(NTHREADS)
void k_ln2_mlp(const float* __restrict__ xr,
               const float* __restrict__ ln2g,
               const float* __restrict__ ln2b,
               const ushort_t* __restrict__ w1_t,   // [3072 cols][768 K]
               const float* __restrict__ b1,
               const ushort_t* __restrict__ w2_t,   // [768 cols][3072 K]
               const float* __restrict__ b2,
               float* __restrict__ out) {
  extern __shared__ unsigned char smem_raw[];
  float*    red1  = (float*)smem_raw;
  float*    red2  = red1 + NTHREADS;
  float*    muv   = red2 + NTHREADS;
  float*    rsv   = muv + TOK_TILE;
  ushort_t* ybf   = (ushort_t*)(rsv + TOK_TILE);     // [32][768]  bf16 LN2(xr)
  ushort_t* actbf = ybf + TOK_TILE * HDIM;           // [32][3072] bf16 gelu act

  const int tid = threadIdx.x;
  const size_t tile0 = (size_t)blockIdx.x * TOK_TILE;
  const float* xg = xr + tile0 * HDIM;

  // ---- LN2
  {
    int tok = tid >> 3, j = tid & 7;
    const float* xt = xg + (size_t)tok * HDIM;
    float s = 0.f, s2 = 0.f;
    for (int e = j; e < HDIM; e += 8) { float v = xt[e]; s += v; s2 += v * v; }
    red1[tid] = s; red2[tid] = s2;
  }
  __syncthreads();
  if (tid < TOK_TILE) {
    float s = 0.f, s2 = 0.f;
    for (int j = 0; j < 8; j++) { s += red1[tid * 8 + j]; s2 += red2[tid * 8 + j]; }
    float mu  = s * (1.f / HDIM);
    float var = s2 * (1.f / HDIM) - mu * mu;
    muv[tid] = mu;
    rsv[tid] = rsqrtf(var + 1e-6f);
  }
  __syncthreads();
  for (int e = tid; e < TOK_TILE * HDIM; e += NTHREADS) {
    int tok = e / HDIM, h = e - tok * HDIM;
    ybf[e] = f2bf((xg[e] - muv[tok]) * rsv[tok] * ln2g[h] + ln2b[h]);
  }
  __syncthreads();

  const int wave = tid >> 5, lane = tid & 31;
  const int hi = lane >> 4, lm = lane & 15;

  // ---- GEMM1 + exact GELU: 192 col-tiles of 3072, K = 768
  for (int job = wave; job < MDIM / 16; job += 8) {
    int col = job * 16 + lm;
    const ushort_t* bw = w1_t + (size_t)col * HDIM + hi * 16;
    const ushort_t* a0 = ybf + (size_t)lm * HDIM;
    const ushort_t* a1 = ybf + (size_t)(lm + 16) * HDIM;
    v8f acc0 = {}, acc1 = {};
    for (int kk = 0; kk < HDIM; kk += 32) {
      __builtin_prefetch(bw + kk + 128, 0, 0);
      v16bf bfrag = load_b16(bw + kk);
      acc0 = WMMA_BF16(load_a_lds(a0 + kk, hi), bfrag, acc0);
      acc1 = WMMA_BF16(load_a_lds(a1 + kk, hi), bfrag, acc1);
    }
    float bias = b1[col];
    #pragma unroll
    for (int i = 0; i < 8; i++) {
      int r = i + 8 * hi;
      float t0 = acc0[i] + bias, t1 = acc1[i] + bias;
      actbf[(size_t)r * MDIM + col]        = f2bf(0.5f * t0 * (1.f + erff(t0 * 0.70710678f)));
      actbf[(size_t)(r + 16) * MDIM + col] = f2bf(0.5f * t1 * (1.f + erff(t1 * 0.70710678f)));
    }
  }
  __syncthreads();

  // ---- GEMM2 + bias + residual: 48 col-tiles of 768, K = 3072
  for (int job = wave; job < HDIM / 16; job += 8) {
    int col = job * 16 + lm;
    const ushort_t* bw = w2_t + (size_t)col * MDIM + hi * 16;
    const ushort_t* a0 = actbf + (size_t)lm * MDIM;
    const ushort_t* a1 = actbf + (size_t)(lm + 16) * MDIM;
    v8f acc0 = {}, acc1 = {};
    for (int kk = 0; kk < MDIM; kk += 32) {
      __builtin_prefetch(bw + kk + 128, 0, 0);
      v16bf bfrag = load_b16(bw + kk);
      acc0 = WMMA_BF16(load_a_lds(a0 + kk, hi), bfrag, acc0);
      acc1 = WMMA_BF16(load_a_lds(a1 + kk, hi), bfrag, acc1);
    }
    float bias = b2[col];
    #pragma unroll
    for (int i = 0; i < 8; i++) {
      int r = i + 8 * hi;
      size_t g0 = (tile0 + r) * HDIM + col;
      size_t g1 = (tile0 + r + 16) * HDIM + col;
      out[g0] = acc0[i] + bias + xr[g0];
      out[g1] = acc1[i] + bias + xr[g1];
    }
  }
}

// ---------------------------------------------------------------------------
extern "C" void kernel_launch(void* const* d_in, const int* in_sizes, int n_in,
                              void* d_out, int out_size, void* d_ws, size_t ws_size,
                              hipStream_t stream) {
  (void)in_sizes; (void)n_in; (void)out_size; (void)ws_size;

  const float* x     = (const float*)d_in[0];
  const float* W_qkv = (const float*)d_in[1];
  const float* b_qkv = (const float*)d_in[2];
  const float* W_out = (const float*)d_in[3];
  const float* b_out = (const float*)d_in[4];
  const float* ln1g  = (const float*)d_in[5];
  const float* ln1b  = (const float*)d_in[6];
  const float* ln2g  = (const float*)d_in[7];
  const float* ln2b  = (const float*)d_in[8];
  const float* W1    = (const float*)d_in[9];
  const float* b1    = (const float*)d_in[10];
  const float* W2    = (const float*)d_in[11];
  const float* b2    = (const float*)d_in[12];
  float* out = (float*)d_out;

  // workspace layout: bf16 transposed weights (L2-resident, ~10.6 MB) + xr f32
  unsigned char* ws = (unsigned char*)d_ws;
  size_t off = 0;
  ushort_t* wqkv_t = (ushort_t*)(ws + off); off += (size_t)3 * HDIM * KFDIM * 2;
  ushort_t* wout_t = (ushort_t*)(ws + off); off += (size_t)KFDIM * KFDIM * 2;
  ushort_t* w1_t   = (ushort_t*)(ws + off); off += (size_t)MDIM * HDIM * 2;
  ushort_t* w2_t   = (ushort_t*)(ws + off); off += (size_t)HDIM * MDIM * 2;
  off = (off + 255) & ~(size_t)255;
  float* xr = (float*)(ws + off);

  k_convert_transpose<<<dim3(96, 3), NTHREADS, 0, stream>>>(W_qkv, wqkv_t, KFDIM, 3 * KFDIM);
  k_convert_transpose<<<dim3(64, 1), NTHREADS, 0, stream>>>(W_out, wout_t, KFDIM, KFDIM);
  k_convert_transpose<<<dim3(512, 1), NTHREADS, 0, stream>>>(W1, w1_t, HDIM, MDIM);
  k_convert_transpose<<<dim3(512, 1), NTHREADS, 0, stream>>>(W2, w2_t, MDIM, HDIM);

  const size_t smem1 = (size_t)(NTHREADS * 2 + TOK_TILE * 2) * sizeof(float)
                     + (size_t)TOK_TILE * (HDIM + QKVDIM + HDIM) * sizeof(ushort_t);
  const size_t smem2 = (size_t)(NTHREADS * 2 + TOK_TILE * 2) * sizeof(float)
                     + (size_t)TOK_TILE * (HDIM + MDIM) * sizeof(ushort_t);

  k_ln1_qkv_attn<<<TTOK / TOK_TILE, NTHREADS, smem1, stream>>>(
      x, b_qkv, b_out, ln1g, ln1b, wqkv_t, wout_t, xr);
  k_ln2_mlp<<<TTOK / TOK_TILE, NTHREADS, smem2, stream>>>(
      xr, ln2g, ln2b, w1_t, b1, w2_t, b2, out);
}
